// SiameseGNN_14937896255716
// MI455X (gfx1250) — compile-verified, matrix operand order checked by reading, not run on hardware
//
#include <hip/hip_runtime.h>

#define NN 50000
#define NE 1600000

typedef __attribute__((ext_vector_type(16))) _Float16 v16h;
typedef __attribute__((ext_vector_type(8)))  float    v8f;

// K-offset map for 16-bit WMMA A/B fragments (ISA 7.12.2, 16-bit 16x32):
// lane<16 : halves 0..7 -> K 0..7,  halves 8..15 -> K 16..23
// lane>=16: halves 0..7 -> K 8..15, halves 8..15 -> K 24..31
__device__ __forceinline__ int koff_map(int lane, int h) {
    int base = (h < 8) ? h : (h + 8);
    return (lane < 16) ? base : (base + 8);
}

// ---------------- degree ----------------
__global__ __launch_bounds__(256) void deg_kernel(const int* __restrict__ dst,
                                                  float* __restrict__ deg) {
    int e = blockIdx.x * 256 + threadIdx.x;
    if (e < NE) atomicAdd(&deg[dst[e]], 1.0f);
}

__global__ __launch_bounds__(256) void invdeg_kernel(float* __restrict__ deg) {
    int i = blockIdx.x * 256 + threadIdx.x;
    if (i < NN) deg[i] = 1.0f / fmaxf(deg[i], 1.0f);
}

// ---------------- scatter-add of 128-wide rows (mean-agg numerator) --------
// one thread handles 4 consecutive floats of one edge (32 threads per edge)
__global__ __launch_bounds__(256) void scatter_kernel(const float* __restrict__ x,
                                                      const int* __restrict__ src,
                                                      const int* __restrict__ dst,
                                                      float* __restrict__ agg) {
    unsigned t = blockIdx.x * 256u + threadIdx.x;
    unsigned e = t >> 5;
    unsigned part = t & 31u;
    if (e < NE) {
        int s = src[e];
        int d = dst[e];
        const float4 v = *(const float4*)(x + (size_t)s * 128 + part * 4);
        float* o = agg + (size_t)d * 128 + part * 4;
        atomicAdd(o + 0, v.x);
        atomicAdd(o + 1, v.y);
        atomicAdd(o + 2, v.z);
        atomicAdd(o + 3, v.w);
    }
}

// ---------------- pack activations into WMMA A-fragment layout -------------
// A part (widthA cols, optionally row-scaled) concatenated with B part.
__global__ __launch_bounds__(256) void packA_kernel(const float* __restrict__ A,
                                                    const float* __restrict__ B,
                                                    const float* __restrict__ rowscale,
                                                    int widthA, int widthB,
                                                    v16h* __restrict__ out) {
    int Ktiles = (widthA + widthB) >> 5;
    int total = (NN / 16) * Ktiles * 32;
    int t = blockIdx.x * 256 + threadIdx.x;
    if (t >= total) return;
    int lane = t & 31;
    int kb = (t >> 5) % Ktiles;
    int mt = (t >> 5) / Ktiles;
    int row = mt * 16 + (lane & 15);
    float scale = rowscale ? rowscale[row] : 1.0f;
    v16h frag;
    for (int h = 0; h < 16; ++h) {
        int k = kb * 32 + koff_map(lane, h);
        float v;
        if (k < widthA) v = A[(size_t)row * widthA + k] * scale;
        else            v = B[(size_t)row * widthB + (k - widthA)];
        frag[h] = (_Float16)v;
    }
    out[t] = frag;  // index = (mt*Ktiles + kb)*32 + lane
}

// ---------------- pack weights into WMMA B-fragment layout -----------------
// W1 rows [0,ksplit), W2 rows [ksplit,K); both row-major [fan_in, fan_out].
__global__ __launch_bounds__(256) void packB_kernel(const float* __restrict__ W1,
                                                    const float* __restrict__ W2,
                                                    int ksplit, int K, int Nout,
                                                    v16h* __restrict__ out) {
    int Ktiles = K >> 5;
    int Ntiles = Nout >> 4;
    int total = Ktiles * Ntiles * 32;
    int t = blockIdx.x * 256 + threadIdx.x;
    if (t >= total) return;
    int lane = t & 31;
    int nb = (t >> 5) % Ntiles;
    int kb = (t >> 5) / Ntiles;
    int col = nb * 16 + (lane & 15);
    v16h frag;
    for (int h = 0; h < 16; ++h) {
        int k = kb * 32 + koff_map(lane, h);
        float v = (k < ksplit) ? W1[(size_t)k * Nout + col]
                               : W2[(size_t)(k - ksplit) * Nout + col];
        frag[h] = (_Float16)v;
    }
    out[t] = frag;  // index = (kb*Ntiles + nb)*32 + lane
}

// ---------------- WMMA GEMM: [NN, 32*Kt](f16 frags) @ [32*Kt, 16*Nt] -------
// one block per 16-row tile. A fragments are staged into LDS with CDNA5
// async global->LDS copies (ASYNCcnt-tracked, no VGPR round trip); all waves
// of the block then reuse them from LDS. Wave w computes output col-tile w.
__global__ __launch_bounds__(256) void wmma_gemm_kernel(const v16h* __restrict__ Afrag,
                                                        const v16h* __restrict__ Bfrag,
                                                        const float* __restrict__ bias,
                                                        float* __restrict__ out,
                                                        int Ktiles, int Ntiles, int relu) {
    __shared__ _Float16 ldsA[8 * 32 * 16];  // up to 8 K-tiles, 8 KB
    int lane = threadIdx.x & 31;
    int wave = threadIdx.x >> 5;
    int nwaves = blockDim.x >> 5;
    int mt = blockIdx.x;

    // async copy: each lane moves its 32-byte A fragment straight into LDS.
    // INST_OFFSET is applied to both LDS and global addresses (ISA 10.7),
    // so offset:16 moves the second 16B half correctly on both sides.
    // The LDS offset is derived from the real array address (low 32 bits of
    // the LDS-aperture flat address = wave-relative LDS byte address); the
    // ptrtoint escape + "memory" clobber force the compiler to assume the
    // asm writes ldsA, keeping the ds_loads below alive.
    for (int kb = wave; kb < Ktiles; kb += nwaves) {
        unsigned long long ga =
            (unsigned long long)&Afrag[((size_t)mt * Ktiles + kb) * 32 + lane];
        unsigned lds_off =
            (unsigned)(unsigned long long)&ldsA[(kb * 32 + lane) * 16];
        asm volatile(
            "global_load_async_to_lds_b128 %[lds], %[ga], off\n\t"
            "global_load_async_to_lds_b128 %[lds], %[ga], off offset:16"
            :
            : [lds] "v"(lds_off), [ga] "v"(ga)
            : "memory");
    }
    {
        unsigned lds_base = (unsigned)(unsigned long long)&ldsA[0];
        asm volatile("s_wait_asynccnt 0" :: "v"(lds_base) : "memory");
    }
    __syncthreads();

    int nt = wave;
    v8f c = {0.f, 0.f, 0.f, 0.f, 0.f, 0.f, 0.f, 0.f};
    for (int kb = 0; kb < Ktiles; ++kb) {
        v16h a = *(const v16h*)&ldsA[(kb * 32 + lane) * 16];
        v16h b = Bfrag[((size_t)kb * Ntiles + nt) * 32 + lane];
        c = __builtin_amdgcn_wmma_f32_16x16x32_f16(false, a, false, b,
                                                   (short)0, c, false, false);
    }

    int ldo = Ntiles * 16;
    int col = nt * 16 + (lane & 15);
    float bv = bias[col];
    int rowbase = mt * 16 + ((lane >= 16) ? 8 : 0);
    for (int v = 0; v < 8; ++v) {
        float val = c[v] + bv;
        if (relu) val = fmaxf(val, 0.0f);
        out[(size_t)(rowbase + v) * ldo + col] = val;
    }
}

// ---------------- row-wise L2 normalize ------------------------------------
__global__ __launch_bounds__(256) void l2norm_kernel(const float* __restrict__ in,
                                                     float* __restrict__ out, int W) {
    int i = blockIdx.x * 256 + threadIdx.x;
    if (i >= NN) return;
    const float* r = in + (size_t)i * W;
    float ss = 0.f;
    for (int j = 0; j < W; j += 4) {
        float4 v = *(const float4*)(r + j);
        ss += v.x * v.x + v.y * v.y + v.z * v.z + v.w * v.w;
    }
    float scale = 1.0f / fmaxf(sqrtf(ss), 1e-12f);
    float* o = out + (size_t)i * W;
    for (int j = 0; j < W; j += 4) {
        float4 v = *(const float4*)(r + j);
        float4 s = {v.x * scale, v.y * scale, v.z * scale, v.w * scale};
        *(float4*)(o + j) = s;
    }
}

// ---------------- final 64 -> 1 head ---------------------------------------
__global__ __launch_bounds__(256) void nulldot_kernel(const float* __restrict__ q,
                                                      const float* __restrict__ w,
                                                      const float* __restrict__ b,
                                                      float* __restrict__ out) {
    int i = blockIdx.x * 256 + threadIdx.x;
    if (i >= NN) return;
    float s = b[0];
    const float* r = q + (size_t)i * 64;
    for (int j = 0; j < 64; ++j) s += r[j] * w[j];
    out[i] = s;
}

static inline unsigned cdivu(size_t a, size_t b) { return (unsigned)((a + b - 1) / b); }

extern "C" void kernel_launch(void* const* d_in, const int* in_sizes, int n_in,
                              void* d_out, int out_size, void* d_ws, size_t ws_size,
                              hipStream_t stream) {
    (void)in_sizes; (void)n_in; (void)out_size; (void)ws_size;

    const float* xA  = (const float*)d_in[0];
    const int*   eiA = (const int*)d_in[1];
    const float* xB  = (const float*)d_in[2];
    const int*   eiB = (const int*)d_in[3];
    const float* W1l = (const float*)d_in[4];
    const float* b1  = (const float*)d_in[5];
    const float* W1r = (const float*)d_in[6];
    const float* W2l = (const float*)d_in[7];
    const float* b2  = (const float*)d_in[8];
    const float* W2r = (const float*)d_in[9];
    const float* W3l = (const float*)d_in[10];
    const float* b3  = (const float*)d_in[11];
    const float* W3r = (const float*)d_in[12];
    const float* Wp1 = (const float*)d_in[13];
    const float* bp1 = (const float*)d_in[14];
    const float* Wp2 = (const float*)d_in[15];
    const float* bp2 = (const float*)d_in[16];
    const float* Wn1 = (const float*)d_in[17];
    const float* bn1 = (const float*)d_in[18];
    const float* Wn2 = (const float*)d_in[19];
    const float* bn2 = (const float*)d_in[20];

    // ---- workspace carve-up ----
    char* wsp = (char*)d_ws;
    auto carve = [&](size_t bytes) {
        void* p = wsp;
        wsp += (bytes + 255) & ~(size_t)255;
        return p;
    };
    float* deg   = (float*)carve((size_t)NN * 4);
    float* agg   = (float*)carve((size_t)NN * 128 * 4);
    float* bufA  = (float*)carve((size_t)NN * 128 * 4);
    float* bufB  = (float*)carve((size_t)NN * 128 * 4);
    v16h*  afrag = (v16h*)carve((size_t)NN * 256 * 2);
    float* p1    = (float*)carve((size_t)NN * 64 * 4);
    float* q1    = (float*)carve((size_t)NN * 64 * 4);
    v16h*  wc1   = (v16h*)carve((size_t)256 * 128 * 2);
    v16h*  wc2   = (v16h*)carve((size_t)256 * 128 * 2);
    v16h*  wc3   = (v16h*)carve((size_t)256 * 128 * 2);
    v16h*  wp1f  = (v16h*)carve((size_t)128 * 64 * 2);
    v16h*  wp2f  = (v16h*)carve((size_t)64 * 64 * 2);
    v16h*  wn1f  = (v16h*)carve((size_t)192 * 64 * 2);

    float* out   = (float*)d_out;
    float* outHA = out;
    float* outHB = out + (size_t)NN * 128;
    float* outZA = out + (size_t)2 * NN * 128;
    float* outZB = outZA + (size_t)NN * 64;
    float* outNA = outZB + (size_t)NN * 64;
    float* outNB = outNA + NN;

    // ---- pack weights once (tiny) ----
    packB_kernel<<<cdivu(8 * 8 * 32, 256), 256, 0, stream>>>(W1l, W1r, 128, 256, 128, wc1);
    packB_kernel<<<cdivu(8 * 8 * 32, 256), 256, 0, stream>>>(W2l, W2r, 128, 256, 128, wc2);
    packB_kernel<<<cdivu(8 * 8 * 32, 256), 256, 0, stream>>>(W3l, W3r, 128, 256, 128, wc3);
    packB_kernel<<<cdivu(4 * 4 * 32, 256), 256, 0, stream>>>(Wp1, Wp1, 128, 128, 64, wp1f);
    packB_kernel<<<cdivu(2 * 4 * 32, 256), 256, 0, stream>>>(Wp2, Wp2, 64, 64, 64, wp2f);
    packB_kernel<<<cdivu(6 * 4 * 32, 256), 256, 0, stream>>>(Wn1, Wn1, 192, 192, 64, wn1f);

    auto run_graph = [&](const float* x, const int* ei,
                         float* outH, float* outZ, float* outNull) {
        const int* srcp = ei;
        const int* dstp = ei + NE;

        (void)hipMemsetAsync(deg, 0, (size_t)NN * 4, stream);
        deg_kernel<<<cdivu(NE, 256), 256, 0, stream>>>(dstp, deg);
        invdeg_kernel<<<cdivu(NN, 256), 256, 0, stream>>>(deg);

        auto sage_layer = [&](const float* xin, const v16h* wfrag, const float* bias,
                              float* hout, int relu) {
            (void)hipMemsetAsync(agg, 0, (size_t)NN * 128 * 4, stream);
            scatter_kernel<<<NE / 8, 256, 0, stream>>>(xin, srcp, dstp, agg);
            packA_kernel<<<cdivu((size_t)(NN / 16) * 8 * 32, 256), 256, 0, stream>>>(
                agg, xin, deg, 128, 128, afrag);
            wmma_gemm_kernel<<<NN / 16, 256, 0, stream>>>(afrag, wfrag, bias, hout,
                                                          8, 8, relu);
        };
        sage_layer(x, wc1, b1, bufA, 1);      // h1
        sage_layer(bufA, wc2, b2, bufB, 1);   // h2
        sage_layer(bufB, wc3, b3, bufA, 0);   // h3 (kept for heads)

        l2norm_kernel<<<cdivu(NN, 256), 256, 0, stream>>>(bufA, outH, 128);

        // projection head: relu(h3@Wp1+bp1) @ Wp2 + bp2, then L2 norm -> z
        packA_kernel<<<cdivu((size_t)(NN / 16) * 4 * 32, 256), 256, 0, stream>>>(
            bufA, nullptr, nullptr, 128, 0, afrag);
        wmma_gemm_kernel<<<NN / 16, 128, 0, stream>>>(afrag, wp1f, bp1, p1, 4, 4, 1);
        packA_kernel<<<cdivu((size_t)(NN / 16) * 2 * 32, 256), 256, 0, stream>>>(
            p1, nullptr, nullptr, 64, 0, afrag);
        wmma_gemm_kernel<<<NN / 16, 128, 0, stream>>>(afrag, wp2f, bp2, q1, 2, 4, 0);
        l2norm_kernel<<<cdivu(NN, 256), 256, 0, stream>>>(q1, outZ, 64);

        // null head: relu(concat(h3, z) @ Wn1 + bn1) @ Wn2 + bn2
        packA_kernel<<<cdivu((size_t)(NN / 16) * 6 * 32, 256), 256, 0, stream>>>(
            bufA, outZ, nullptr, 128, 64, afrag);
        wmma_gemm_kernel<<<NN / 16, 128, 0, stream>>>(afrag, wn1f, bn1, q1, 6, 4, 1);
        nulldot_kernel<<<cdivu(NN, 256), 256, 0, stream>>>(q1, Wn2, bn2, outNull);
    };

    run_graph(xA, eiA, outHA, outZA, outNA);
    run_graph(xB, eiB, outHB, outZB, outNB);
}
